// CBConv2d_65111704207914
// MI455X (gfx1250) — matrix operand order repaired and checked by reading, not compiled, not run on hardware
//
#include <hip/hip_runtime.h>
#include <hip/hip_bf16.h>

// CDNA5 (gfx1250) change-based conv2d, fully fused:
//   out = where(dilate3x3(any_c(|inp - prev_in| > 0.1)), conv3x3(inp)+bias, prev_out)
//
// Single WMMA conv kernel fuses change detection + dilation into its LDS
// staging pass (the staged 3x66 halo tile IS the dilation receptive field).
// HBM traffic ~= inp + prev_in + prev_out + out = 256MB -> ~11us @ 23.3 TB/s.
// 19.3 GFLOP matmul runs on v_wmma_f32_16x16x32_f16 (f16 in, f32 acc).

typedef _Float16 v8h  __attribute__((ext_vector_type(8)));
typedef _Float16 v16h __attribute__((ext_vector_type(16)));
typedef float    v8f  __attribute__((ext_vector_type(8)));

#define IMG_H 512
#define IMG_W 512
#define NCH   64
#define HWPIX (IMG_H * IMG_W)
#define NTAPS 9
#define KPERC (NTAPS * NCH)      // 576 = K dimension
#define LDS_STRIDE 68            // 64 cin + 4 pad halves (bank-conflict-free fill)
#define TILE_COLS 66             // 64 pixels + 2 halo
#define TILE_POS  (3 * TILE_COLS)
#define TILE_BYTES (TILE_POS * LDS_STRIDE * 2)

// ---- kernel 1: f32 weights [cout][cin][kh][kw] -> f16 [cout][tap][cin] ----
__global__ __launch_bounds__(256) void weight_cvt_kernel(
    const float* __restrict__ w, _Float16* __restrict__ w16) {
  int i = blockIdx.x * 256 + threadIdx.x;
  if (i >= NCH * KPERC) return;
  int cout = i / KPERC;
  int rem  = i % KPERC;
  int tap  = rem >> 6;
  int cin  = rem & 63;
  w16[i] = (_Float16)w[(cout * NCH + cin) * NTAPS + tap];
}

// ---- kernel 2: fused change-detect + dilate + implicit-GEMM WMMA conv ----
// Block: 256 threads = 8 wave32. Tile: one image row h, 64 pixels, all 64 couts.
// GEMM: M=64 (cout) x N=64 (pixel) x K=576, K-chunk = (tap, cin-half of 32).
__global__ __launch_bounds__(256) void conv_wmma_kernel(
    const float* __restrict__ inp, const float* __restrict__ prev_in,
    const float* __restrict__ prev_out,
    const _Float16* __restrict__ w16, const float* __restrict__ bias,
    float* __restrict__ out) {
  extern __shared__ char smem[];
  _Float16* lds = (_Float16*)smem;                      // [3*66][stride 68] f16
  unsigned char* chgT = (unsigned char*)(smem + TILE_BYTES); // [198] changed
  unsigned char* dilT = chgT + TILE_POS;                     // [64] dilated

  const int tid = threadIdx.x;
  const int bid = blockIdx.x;
  const int h   = bid >> 3;           // image row
  const int w0  = (bid & 7) << 6;     // first of 64 output pixels

  // Stage input tile (3 rows x 66 cols x 64 ch) into LDS as f16, zero-padded,
  // while computing the per-position change flag (any channel delta > 0.1).
  for (int pos = tid; pos < TILE_POS; pos += 256) {
    const int r = pos / TILE_COLS, c = pos % TILE_COLS;
    const int gh = h + r - 1, gw = w0 + c - 1;
    _Float16* dst = &lds[pos * LDS_STRIDE];
    unsigned flag = 0;
    if ((unsigned)gh < IMG_H && (unsigned)gw < IMG_W) {
      const float* src  = inp     + gh * IMG_W + gw;
      const float* psrc = prev_in + gh * IMG_W + gw;
      #pragma unroll 8
      for (int ch = 0; ch < NCH; ++ch) {
        float a = src[ch * HWPIX];
        float b = psrc[ch * HWPIX];
        dst[ch] = (_Float16)a;
        flag |= (fabsf(a - b) > 0.1f) ? 1u : 0u;
      }
    } else {
      #pragma unroll 8
      for (int ch = 0; ch < NCH; ++ch)
        dst[ch] = (_Float16)0.0f;
    }
    chgT[pos] = (unsigned char)flag;
  }
  __syncthreads();

  // 3x3 dilation of the change map; neighborhood of output pixel (tile col
  // c = pix+1) is rows 0..2, cols pix..pix+2 -- entirely inside the halo tile.
  if (tid < 64) {
    unsigned v = 0;
    #pragma unroll
    for (int r = 0; r < 3; ++r)
      #pragma unroll
      for (int dc = 0; dc < 3; ++dc)
        v |= chgT[r * TILE_COLS + tid + dc];
    dilT[tid] = (unsigned char)v;
  }
  __syncthreads();

  const int lane = tid & 31;
  const int wave = tid >> 5;
  const int mt   = wave & 3;            // cout tile (16 couts)
  const int nt0  = (wave >> 2) << 1;    // this wave does pixel tiles nt0, nt0+1
  const int ln   = lane & 15;
  const int hi   = lane >> 4;           // lane half; selects K-half 16*hi (this code's K order)

  v8f acc0 = {}; v8f acc1 = {};
  // A source: w16[cout = mt*16+ln][tap][cin]; lane<16 holds K 0-15,
  // lane>=16 holds K 16-31 (matches B) -> two contiguous v8h loads per chunk.
  const _Float16* wbase = w16 + (mt * 16 + ln) * KPERC;

  #pragma unroll
  for (int tap = 0; tap < NTAPS; ++tap) {
    const int kh = tap / 3, kw = tap % 3;
    const _Float16* lrow = &lds[(kh * TILE_COLS + kw) * LDS_STRIDE];
    const _Float16* wtap = wbase + tap * NCH;
    #pragma unroll
    for (int cb = 0; cb < NCH; cb += 32) {
      // A fragment (16x32 f16): K = cb + hi*16 .. hi*16+15
      v8h alo = *(const v8h*)(wtap + cb + hi * 16);
      v8h ahi = *(const v8h*)(wtap + cb + hi * 16 + 8);
      v16h a = __builtin_shufflevector(alo, ahi,
               0,1,2,3,4,5,6,7,8,9,10,11,12,13,14,15);
      // B fragments (32x16 f16): lane = column, 16 contiguous cin halves
      const int cidx = cb + hi * 16;
      v16h b0 = *(const v16h*)(lrow + (nt0 * 16 + ln) * LDS_STRIDE + cidx);
      v16h b1 = *(const v16h*)(lrow + ((nt0 + 1) * 16 + ln) * LDS_STRIDE + cidx);
      acc0 = __builtin_amdgcn_wmma_f32_16x16x32_f16(
          false, a, false, b0, (short)0, acc0, false, false);
      acc1 = __builtin_amdgcn_wmma_f32_16x16x32_f16(
          false, a, false, b1, (short)0, acc1, false, false);
    }
  }

  // Store: C/D layout -> VGPR r holds M = r + 8*hi, N = lane&15.
  const int coutBase = mt * 16 + hi * 8;
  #pragma unroll
  for (int t = 0; t < 2; ++t) {
    const v8f acc = t ? acc1 : acc0;
    const int pix = (nt0 + t) * 16 + ln;
    const int wpix = w0 + pix;
    const bool changed = dilT[pix] != 0;
    #pragma unroll
    for (int r = 0; r < 8; ++r) {
      const int cout = coutBase + r;
      const long idx = ((long)cout * IMG_H + h) * IMG_W + wpix;
      out[idx] = changed ? (acc[r] + bias[cout]) : prev_out[idx];
    }
  }
}

extern "C" void kernel_launch(void* const* d_in, const int* in_sizes, int n_in,
                              void* d_out, int out_size, void* d_ws, size_t ws_size,
                              hipStream_t stream) {
  const float* inp      = (const float*)d_in[0];
  const float* prev_in  = (const float*)d_in[1];
  const float* prev_out = (const float*)d_in[2];
  const float* weight   = (const float*)d_in[3];
  const float* bias     = (const float*)d_in[4];
  float* out = (float*)d_out;

  // workspace: f16 weights [cout][tap][cin], 72KB
  _Float16* w16 = (_Float16*)d_ws;

  weight_cvt_kernel<<<(NCH * KPERC + 255) / 256, 256, 0, stream>>>(weight, w16);

  const size_t ldsBytes = TILE_BYTES + TILE_POS + 64 + 2; // tile + chg + dil
  conv_wmma_kernel<<<IMG_H * (IMG_W / 64), 256, ldsBytes, stream>>>(
      inp, prev_in, prev_out, w16, bias, out);
}
